// Quantizer_39797166965033
// MI455X (gfx1250) — matrix operand order repaired and checked by reading, not compiled, not run on hardware
//
#include <hip/hip_runtime.h>
#include <hip/hip_bf16.h>
#include <math.h>

typedef __attribute__((ext_vector_type(16))) _Float16 v16h;
typedef __attribute__((ext_vector_type(8)))  float    v8f;
typedef __attribute__((ext_vector_type(4)))  _Float16 v4h;

namespace {
constexpr int    kNC = 4, kM = 512, kD = 64, kN = 65536;
constexpr int    kRowStr  = 72;                       // halves per LDS codebook row (144 B, pad vs bank conflicts)
constexpr size_t kQElems  = (size_t)kN * kNC * kD;    // 16777216 floats of quantized_st
constexpr size_t kScalar0 = kQElems;                  // commitment, codebook, perplexity
constexpr size_t kIdx0    = kQElems + 3;              // indices (N, NC) as float
constexpr int    kLdsB    = kM * kRowStr * 2;         // 73728 B of f16 codebook
constexpr int    kLdsE2   = kLdsB;                    // 512 floats of e^2
constexpr int    kLdsIdx  = kLdsB + kM * 4;           // 8 waves * 16 winner indices
constexpr int    kLdsTot  = kLdsIdx + 128 * 4;        // 76288 B (dynamic LDS; WGP has 320 KB)
}

__global__ void vq_zero(int* __restrict__ counts, float* __restrict__ sse) {
    int t = threadIdx.x;
    for (int i = t; i < kNC * kM; i += 256) counts[i] = 0;
    if (t == 0) *sse = 0.0f;
}

__global__ __launch_bounds__(256)
void vq_main(const float* __restrict__ x, const float* __restrict__ emb,
             const float* __restrict__ mask, float* __restrict__ out,
             int* __restrict__ counts, float* __restrict__ sse) {
    extern __shared__ char smem[];
    _Float16* sB   = (_Float16*)smem;
    float*    sE2  = (float*)(smem + kLdsE2);
    int*      sIdx = (int*)(smem + kLdsIdx);

    const int c = blockIdx.y;
    const int t = threadIdx.x;

    // ---- stage codebook c into LDS as f16 (WMMA-friendly row-major), compute e^2 ----
    for (int row = t; row < kM; row += 256) {
        const float4* er = (const float4*)(emb + (size_t)(c * kM + row) * kD);
        _Float16* dst = sB + row * kRowStr;
        float s = 0.0f;
        #pragma unroll
        for (int j = 0; j < 16; ++j) {
            float4 v = er[j];
            s += v.x * v.x + v.y * v.y + v.z * v.z + v.w * v.w;
            v4h h; h[0] = (_Float16)v.x; h[1] = (_Float16)v.y;
                   h[2] = (_Float16)v.z; h[3] = (_Float16)v.w;
            *(v4h*)(dst + j * 4) = h;
        }
        sE2[row] = s;
    }
    __syncthreads();

    const int w  = t >> 5;       // wave in block
    const int l  = t & 31;       // lane
    const int lr = l & 15;       // row-in-tile (A) / col-in-tile (B,C)
    const int hi = l >> 4;       // lane half
    const int n0 = (blockIdx.x * 8 + w) * 16;
    const int arow = n0 + lr;

    // ---- load A = 16 x-rows, f32 -> f16 in the A-operand lane layout ----
    // half 0: K {0..7, 16..23} (+32*kc) ; half 1: K {8..15, 24..31} (+32*kc)
    const float* xr = x + (size_t)arow * (kNC * kD) + c * kD;
    v16h  A[2];
    float fx[2][16];                       // keep exact f32 x for the loss
    #pragma unroll
    for (int kc = 0; kc < 2; ++kc) {
        const float* p = xr + kc * 32 + hi * 8;
        float4 f0 = *(const float4*)(p);
        float4 f1 = *(const float4*)(p + 4);
        float4 f2 = *(const float4*)(p + 16);
        float4 f3 = *(const float4*)(p + 20);
        float tmp[16] = {f0.x, f0.y, f0.z, f0.w, f1.x, f1.y, f1.z, f1.w,
                         f2.x, f2.y, f2.z, f2.w, f3.x, f3.y, f3.z, f3.w};
        #pragma unroll
        for (int j = 0; j < 16; ++j) { fx[kc][j] = tmp[j]; A[kc][j] = (_Float16)tmp[j]; }
    }

    // ---- sweep all 512 codes: 2 tiles (4 WMMAs) per iteration so tile t+1's
    //      XDL work fills tile t's WMMA->VALU hazard slots and co-executes with
    //      the argmin VALU (16-bit WMMA co-issues like a TRANS op) ----
    float best[8]; int bidx[8];
    #pragma unroll
    for (int v = 0; v < 8; ++v) { best[v] = 3.0e38f; bidx[v] = 0; }

    for (int m0 = 0; m0 < kM; m0 += 32) {
        // B operand: col = lane%16 (= code row in LDS); half 0 -> K {0..7, 16..23}, half 1 -> K {8..15, 24..31}
        const _Float16* bp0 = sB + (m0 + lr) * kRowStr + hi * 8;
        const _Float16* bp1 = bp0 + 16 * kRowStr;
        union { int4 q[2]; v16h h; } B00, B01, B10, B11;
        B00.q[0] = *(const int4*)(bp0);      B00.q[1] = *(const int4*)(bp0 + 16);
        B01.q[0] = *(const int4*)(bp0 + 32); B01.q[1] = *(const int4*)(bp0 + 48);
        B10.q[0] = *(const int4*)(bp1);      B10.q[1] = *(const int4*)(bp1 + 16);
        B11.q[0] = *(const int4*)(bp1 + 32); B11.q[1] = *(const int4*)(bp1 + 48);
        float e20 = sE2[m0 + lr];
        float e21 = sE2[m0 + 16 + lr];
        v8f acc0 = {0.f, 0.f, 0.f, 0.f, 0.f, 0.f, 0.f, 0.f};
        v8f acc1 = {0.f, 0.f, 0.f, 0.f, 0.f, 0.f, 0.f, 0.f};
        acc0 = __builtin_amdgcn_wmma_f32_16x16x32_f16(false, A[0], false, B00.h,
                                                      (short)0, acc0, false, false);
        acc0 = __builtin_amdgcn_wmma_f32_16x16x32_f16(false, A[1], false, B01.h,
                                                      (short)0, acc0, false, false);
        acc1 = __builtin_amdgcn_wmma_f32_16x16x32_f16(false, A[0], false, B10.h,
                                                      (short)0, acc1, false, false);
        acc1 = __builtin_amdgcn_wmma_f32_16x16x32_f16(false, A[1], false, B11.h,
                                                      (short)0, acc1, false, false);
        int mi0 = m0 + lr;
        int mi1 = m0 + 16 + lr;
        #pragma unroll
        for (int v = 0; v < 8; ++v) {
            float sc = e20 - 2.0f * acc0[v];      // rank-equivalent to the distance
            if (sc < best[v]) { best[v] = sc; bidx[v] = mi0; }
        }
        #pragma unroll
        for (int v = 0; v < 8; ++v) {
            float sc = e21 - 2.0f * acc1[v];
            if (sc < best[v]) { best[v] = sc; bidx[v] = mi1; }
        }
    }

    // ---- argmin across the 16 columns of each lane half ----
    #pragma unroll
    for (int v = 0; v < 8; ++v) {
        float bv = best[v]; int bi = bidx[v];
        #pragma unroll
        for (int off = 1; off < 16; off <<= 1) {
            float ov = __shfl_xor(bv, off, 32);
            int   oi = __shfl_xor(bi, off, 32);
            if (ov < bv || (ov == bv && oi < bi)) { bv = ov; bi = oi; }
        }
        best[v] = bv; bidx[v] = bi;
    }
    // lanes 0 / 16 hold winners for rows n0 + hi*8 + v
    if (lr == 0) {
        #pragma unroll
        for (int v = 0; v < 8; ++v) {
            int r = hi * 8 + v;
            sIdx[w * 16 + r] = bidx[v];
            out[kIdx0 + (size_t)(n0 + r) * kNC + c] = (float)bidx[v];
            atomicAdd(counts + c * kM + bidx[v], 1);
        }
    }
    __syncthreads();

    // ---- gather winning codes (exact f32), mask, store quantized, exact SSE ----
    int    qidx = sIdx[w * 16 + lr];
    float  mv   = mask[(size_t)arow * kNC + c];
    const float* eq = emb + (size_t)(c * kM + qidx) * kD;
    float* qo = out + (size_t)arow * (kNC * kD) + c * kD;
    float err = 0.0f;
    #pragma unroll
    for (int kc = 0; kc < 2; ++kc) {
        const float* p = eq + kc * 32 + hi * 8;
        float*       o = qo + kc * 32 + hi * 8;
        float4 g0 = *(const float4*)(p);
        float4 g1 = *(const float4*)(p + 4);
        float4 g2 = *(const float4*)(p + 16);
        float4 g3 = *(const float4*)(p + 20);
        float4 q0 = {g0.x * mv, g0.y * mv, g0.z * mv, g0.w * mv};
        float4 q1 = {g1.x * mv, g1.y * mv, g1.z * mv, g1.w * mv};
        float4 q2 = {g2.x * mv, g2.y * mv, g2.z * mv, g2.w * mv};
        float4 q3 = {g3.x * mv, g3.y * mv, g3.z * mv, g3.w * mv};
        *(float4*)(o)      = q0;
        *(float4*)(o + 4)  = q1;
        *(float4*)(o + 16) = q2;
        *(float4*)(o + 20) = q3;
        float qs[16] = {q0.x, q0.y, q0.z, q0.w, q1.x, q1.y, q1.z, q1.w,
                        q2.x, q2.y, q2.z, q2.w, q3.x, q3.y, q3.z, q3.w};
        #pragma unroll
        for (int j = 0; j < 16; ++j) {
            float d = fx[kc][j] - qs[j];
            err += d * d;
        }
    }
    #pragma unroll
    for (int off = 1; off < 32; off <<= 1) err += __shfl_xor(err, off, 32);
    if (l == 0) atomicAdd(sse, err);
}

__global__ void vq_final(const int* __restrict__ counts, const float* __restrict__ sse,
                         float* __restrict__ out) {
    __shared__ float red[256];
    int t = threadIdx.x;
    float H = 0.0f;
    for (int i = t; i < kNC * kM; i += 256) {
        float p = (float)counts[i] * (1.0f / 65536.0f);
        H += p * logf(p + 1e-10f);
    }
    red[t] = H;
    __syncthreads();
    for (int s = 128; s > 0; s >>= 1) {
        if (t < s) red[t] += red[t + s];
        __syncthreads();
    }
    if (t == 0) {
        float inv = 1.0f / 16777216.0f;   // 1 / (N * NC * D)
        float s = *sse;
        out[kScalar0 + 0] = 0.25f * s * inv;   // commitment_loss
        out[kScalar0 + 1] = s * inv;           // codebook_loss
        out[kScalar0 + 2] = expf(-red[0]);     // perplexity
    }
}

extern "C" void kernel_launch(void* const* d_in, const int* in_sizes, int n_in,
                              void* d_out, int out_size, void* d_ws, size_t ws_size,
                              hipStream_t stream) {
    (void)in_sizes; (void)n_in; (void)out_size; (void)ws_size;
    const float* x    = (const float*)d_in[0];
    const float* emb  = (const float*)d_in[1];
    const float* mask = (const float*)d_in[2];
    float* out = (float*)d_out;
    int*   counts = (int*)d_ws;                       // 2048 ints
    float* sse    = (float*)((char*)d_ws + 8192);     // 1 float

    vq_zero<<<1, 256, 0, stream>>>(counts, sse);
    dim3 grid(kN / 128, kNC);                         // 8 waves x 16 rows per block
    vq_main<<<grid, 256, kLdsTot, stream>>>(x, emb, mask, out, counts, sse);
    vq_final<<<1, 256, 0, stream>>>(counts, sse, out);
}
